// GATModel_12154757448266
// MI455X (gfx1250) — matrix operand order repaired and checked by reading, not compiled, not run on hardware
//
#include <hip/hip_runtime.h>
#include <hip/hip_bf16.h>

// ---------------- problem constants ----------------
#define NN      32768
#define GG      64
#define NPG     512
#define EE      524288
#define F_IN    128
#define HID     128
#define HEADS   2
#define COUT    256      // HEADS*HID
#define KTOP    10
#define OUTC    2
#define EPSLN   1e-5f

typedef __attribute__((ext_vector_type(16))) __bf16 v16bf;
typedef __attribute__((ext_vector_type(8)))  __bf16 v8bf;
typedef __attribute__((ext_vector_type(8)))  float  v8f;

__device__ __forceinline__ float lrelu(float x) { return x > 0.0f ? x : 0.2f * x; }
__device__ __forceinline__ unsigned fkey(float f) {
    int i = __float_as_int(f);
    return (i >= 0) ? ((unsigned)i ^ 0x80000000u) : ~(unsigned)i;
}
__device__ __forceinline__ float funkey(unsigned k) {
    int i = (k & 0x80000000u) ? (int)(k ^ 0x80000000u) : (int)~k;
    return __int_as_float(i);
}

// ---------------- utility kernels ----------------
__global__ void zero_int_kernel(int* p, int n) {
    int i = blockIdx.x * blockDim.x + threadIdx.x;
    if (i < n) p[i] = 0;
}

// split f32 -> (bf16 hi, bf16 lo) elementwise
__global__ void split_kernel(const float* __restrict__ x, int n,
                             __bf16* __restrict__ hi, __bf16* __restrict__ lo) {
    int i = blockIdx.x * blockDim.x + threadIdx.x;
    if (i < n) {
        float v = x[i];
        __bf16 h = (__bf16)v;
        hi[i] = h;
        lo[i] = (__bf16)(v - (float)h);
    }
}

// W[Fin][COUT] row-major -> Wt[COUT][Fin] bf16 hi/lo (transposed for contiguous K loads)
__global__ void wsplit_kernel(const float* __restrict__ W, int Fin,
                              __bf16* __restrict__ Wth, __bf16* __restrict__ Wtl) {
    int i = blockIdx.x * blockDim.x + threadIdx.x;
    if (i < Fin * COUT) {
        int fi = i / COUT, co = i % COUT;
        float v = W[i];
        __bf16 h = (__bf16)v;
        Wth[(size_t)co * Fin + fi] = h;
        Wtl[(size_t)co * Fin + fi] = (__bf16)(v - (float)h);
    }
}

// ---------------- WMMA GEMM: H[N][256] = X[N][Fin] @ W[Fin][256] ----------------
// bf16x3 split precision. One 16x16 output tile per wave. Exact grid (no divergence).
__global__ void gemm_wmma_kernel(const __bf16* __restrict__ Xhi, const __bf16* __restrict__ Xlo,
                                 const __bf16* __restrict__ Wth, const __bf16* __restrict__ Wtl,
                                 float* __restrict__ H, int Fin) {
    int wave = (int)((blockIdx.x * blockDim.x + threadIdx.x) >> 5);
    int lane = threadIdx.x & 31;
    const int tilesN = COUT / 16;          // 16
    int tileN = wave % tilesN;
    int tileM = wave / tilesN;
    int l15 = lane & 15;
    int hiHalf = (lane >> 4) & 1;
    int row = tileM * 16 + l15;
    int col = tileN * 16 + l15;

    const __bf16* xh = Xhi + (size_t)row * Fin;
    const __bf16* xl = Xlo + (size_t)row * Fin;
    const __bf16* wh = Wth + (size_t)col * Fin;
    const __bf16* wl = Wtl + (size_t)col * Fin;
    // ISA 16-bit A layout: lane<16 holds K[0..7],K[16..23]; lane>=16 holds K[8..15],K[24..31]
    int ka0 = hiHalf * 8;
    int ka1 = ka0 + 16;
    // ISA 16-bit B layout: lane<16 holds K[0..7],K[16..23]; lane>=16 holds K[8..15],K[24..31]
    int kb0 = hiHalf * 8;
    int kb1 = kb0 + 16;

    v8f acc = {};
    for (int k = 0; k < Fin; k += 32) {
        union { v16bf v; v8bf h[2]; } ah, al, bh, bl;
        ah.h[0] = *(const v8bf*)(xh + k + ka0);
        ah.h[1] = *(const v8bf*)(xh + k + ka1);
        al.h[0] = *(const v8bf*)(xl + k + ka0);
        al.h[1] = *(const v8bf*)(xl + k + ka1);
        bh.h[0] = *(const v8bf*)(wh + k + kb0);
        bh.h[1] = *(const v8bf*)(wh + k + kb1);
        bl.h[0] = *(const v8bf*)(wl + k + kb0);
        bl.h[1] = *(const v8bf*)(wl + k + kb1);
        acc = __builtin_amdgcn_wmma_f32_16x16x32_bf16(false, ah.v, false, bh.v, (short)0, acc, false, false);
        acc = __builtin_amdgcn_wmma_f32_16x16x32_bf16(false, ah.v, false, bl.v, (short)0, acc, false, false);
        acc = __builtin_amdgcn_wmma_f32_16x16x32_bf16(false, al.v, false, bh.v, (short)0, acc, false, false);
    }
    // D layout: VGPR i -> M = i + 8*hiHalf, N = lane&15
    int rbase = tileM * 16 + hiHalf * 8;
#pragma unroll
    for (int i = 0; i < 8; ++i)
        H[(size_t)(rbase + i) * COUT + col] = acc[i];
}

// ---------------- CSR build (once; graph is static across layers) ----------------
__global__ void hist_kernel(const int* __restrict__ dst, int* __restrict__ counts) {
    int e = blockIdx.x * blockDim.x + threadIdx.x;
    if (e < EE) atomicAdd(&counts[dst[e]], 1);
}
__global__ void scan1_kernel(const int* __restrict__ counts, int* __restrict__ offs,
                             int* __restrict__ bsums) {
    __shared__ int tmp[1024];
    int t = threadIdx.x, b = blockIdx.x;
    int gi = b * 1024 + t;
    int v = counts[gi];
    tmp[t] = v;
    __syncthreads();
    for (int o = 1; o < 1024; o <<= 1) {
        int add = (t >= o) ? tmp[t - o] : 0;
        __syncthreads();
        tmp[t] += add;
        __syncthreads();
    }
    offs[gi] = tmp[t] - v;                 // exclusive
    if (t == 1023) bsums[b] = tmp[t];
}
__global__ void scan2_kernel(int* __restrict__ bsums, int nb) {
    if (threadIdx.x == 0) {
        int acc = 0;
        for (int i = 0; i < nb; ++i) { int v = bsums[i]; bsums[i] = acc; acc += v; }
    }
}
__global__ void scan3_kernel(int* __restrict__ offs, const int* __restrict__ bsums) {
    int t = threadIdx.x, b = blockIdx.x;
    int gi = b * 1024 + t;
    offs[gi] += bsums[b];
    if (b == 0 && t == 0) offs[NN] = EE;
}
__global__ void scatter_kernel(const int* __restrict__ src, const int* __restrict__ dst,
                               const int* __restrict__ offs, int* __restrict__ cursor,
                               int* __restrict__ ssrc) {
    int e = blockIdx.x * blockDim.x + threadIdx.x;
    if (e < EE) {
        int d = dst[e];
        int pos = offs[d] + atomicAdd(&cursor[d], 1);
        ssrc[pos] = src[e];
    }
}

// ---------------- attention ----------------
// per (node,head): ssum = sum_c h*a_src ; dsum = sum_c h*a_dst
__global__ void scores_kernel(const float* __restrict__ H, const float* __restrict__ as,
                              const float* __restrict__ ad, float* __restrict__ ssum,
                              float* __restrict__ dsum) {
    int t = blockIdx.x * blockDim.x + threadIdx.x;
    if (t >= NN * HEADS) return;
    int n = t >> 1, hd = t & 1;
    const float* hr = H + (size_t)n * COUT + hd * HID;
    const float* a = as + hd * HID;
    const float* b = ad + hd * HID;
    float s = 0.0f, d = 0.0f;
    for (int c = 0; c < HID; ++c) { float v = hr[c]; s += v * a[c]; d += v * b[c]; }
    ssum[t] = s; dsum[t] = d;
}
__global__ void node_init_max_kernel(const float* __restrict__ ssum, const float* __restrict__ dsum,
                                     unsigned* __restrict__ mkey) {
    int t = blockIdx.x * blockDim.x + threadIdx.x;
    if (t >= NN * HEADS) return;
    mkey[t] = fkey(lrelu(ssum[t] + dsum[t]));      // self-loop edge
}
__global__ void edge_max_kernel(const int* __restrict__ src, const int* __restrict__ dst,
                                const float* __restrict__ ssum, const float* __restrict__ dsum,
                                unsigned* __restrict__ mkey) {
    int e = blockIdx.x * blockDim.x + threadIdx.x;
    if (e >= EE) return;
    int s = src[e], d = dst[e];
#pragma unroll
    for (int hd = 0; hd < HEADS; ++hd) {
        float v = lrelu(ssum[s * 2 + hd] + dsum[d * 2 + hd]);
        atomicMax(&mkey[d * 2 + hd], fkey(v));
    }
}
__global__ void node_denom_kernel(const float* __restrict__ ssum, const float* __restrict__ dsum,
                                  const unsigned* __restrict__ mkey, float* __restrict__ mf,
                                  float* __restrict__ denom) {
    int t = blockIdx.x * blockDim.x + threadIdx.x;
    if (t >= NN * HEADS) return;
    float m = funkey(mkey[t]);
    mf[t] = m;
    denom[t] = expf(lrelu(ssum[t] + dsum[t]) - m);  // self-loop term
}
__global__ void edge_denom_kernel(const int* __restrict__ src, const int* __restrict__ dst,
                                  const float* __restrict__ ssum, const float* __restrict__ dsum,
                                  const float* __restrict__ mf, float* __restrict__ denom) {
    int e = blockIdx.x * blockDim.x + threadIdx.x;
    if (e >= EE) return;
    int s = src[e], d = dst[e];
#pragma unroll
    for (int hd = 0; hd < HEADS; ++hd) {
        float v = lrelu(ssum[s * 2 + hd] + dsum[d * 2 + hd]);
        atomicAdd(&denom[d * 2 + hd], expf(v - mf[d * 2 + hd]));
    }
}
// one wave per dst node; lane owns channels lane+32*j, j=0..7 (accum in registers)
__global__ void aggregate_kernel(const float* __restrict__ H, const float* __restrict__ ssum,
                                 const float* __restrict__ dsum, const float* __restrict__ mf,
                                 const float* __restrict__ denom, const int* __restrict__ offs,
                                 const int* __restrict__ ssrc, const float* __restrict__ bias,
                                 float* __restrict__ out, int meanHeads) {
    int n = (int)((blockIdx.x * blockDim.x + threadIdx.x) >> 5);
    int lane = threadIdx.x & 31;
    float m0 = mf[n * 2 + 0], m1 = mf[n * 2 + 1];
    float inv0 = 1.0f / denom[n * 2 + 0], inv1 = 1.0f / denom[n * 2 + 1];
    float d0 = dsum[n * 2 + 0], d1 = dsum[n * 2 + 1];
    float acc[8];
#pragma unroll
    for (int j = 0; j < 8; ++j) acc[j] = 0.0f;
    // self loop
    {
        float a0 = expf(lrelu(ssum[n * 2 + 0] + d0) - m0) * inv0;
        float a1 = expf(lrelu(ssum[n * 2 + 1] + d1) - m1) * inv1;
        const float* hr = H + (size_t)n * COUT;
#pragma unroll
        for (int j = 0; j < 8; ++j) { int c = lane + 32 * j; acc[j] += ((c < HID) ? a0 : a1) * hr[c]; }
    }
    int b = offs[n], e2 = offs[n + 1];
    for (int p = b; p < e2; ++p) {
        int s = ssrc[p];
        float a0 = expf(lrelu(ssum[s * 2 + 0] + d0) - m0) * inv0;
        float a1 = expf(lrelu(ssum[s * 2 + 1] + d1) - m1) * inv1;
        const float* hr = H + (size_t)s * COUT;
#pragma unroll
        for (int j = 0; j < 8; ++j) { int c = lane + 32 * j; acc[j] += ((c < HID) ? a0 : a1) * hr[c]; }
    }
    if (!meanHeads) {
#pragma unroll
        for (int j = 0; j < 8; ++j) { int c = lane + 32 * j; out[(size_t)n * COUT + c] = acc[j] + bias[c]; }
    } else {
#pragma unroll
        for (int j = 0; j < 4; ++j) { int c = lane + 32 * j; out[(size_t)n * HID + c] = 0.5f * (acc[j] + acc[j + 4]) + bias[c]; }
    }
}

// ---------------- graph layernorm (scalar moments per graph), optional relu ----------------
__global__ void graph_ln_kernel(float* __restrict__ buf, const float* __restrict__ w,
                                const float* __restrict__ b, int C, int doRelu) {
    __shared__ float red[256];
    int g = blockIdx.x, t = threadIdx.x;
    float* gb = buf + (size_t)g * NPG * C;
    int total = NPG * C;
    float s = 0.0f, sq = 0.0f;
    for (int i = t; i < total; i += 256) { float v = gb[i]; s += v; sq += v * v; }
    red[t] = s; __syncthreads();
    for (int o = 128; o > 0; o >>= 1) { if (t < o) red[t] += red[t + o]; __syncthreads(); }
    float mean = red[0] / (float)total;
    __syncthreads();
    red[t] = sq; __syncthreads();
    for (int o = 128; o > 0; o >>= 1) { if (t < o) red[t] += red[t + o]; __syncthreads(); }
    float var = red[0] / (float)total - mean * mean;
    float rstd = rsqrtf(var + EPSLN);
    int cmask = C - 1;
    for (int i = t; i < total; i += 256) {
        int c = i & cmask;
        float v = (gb[i] - mean) * rstd * w[c] + b[c];
        if (doRelu) v = fmaxf(v, 0.0f);
        gb[i] = v;
    }
}

// ---------------- TopK pooling ----------------
__global__ void pnorm_kernel(const float* __restrict__ p, float* __restrict__ nrm) {
    __shared__ float red[128];
    int t = threadIdx.x;
    float v = p[t];
    red[t] = v * v; __syncthreads();
    for (int o = 64; o > 0; o >>= 1) { if (t < o) red[t] += red[t + o]; __syncthreads(); }
    if (t == 0) nrm[0] = sqrtf(red[0]);
}
__global__ void score_kernel(const float* __restrict__ Hn, const float* __restrict__ p,
                             const float* __restrict__ nrm, float* __restrict__ score) {
    int n = blockIdx.x * blockDim.x + threadIdx.x;
    if (n >= NN) return;
    const float* hr = Hn + (size_t)n * HID;
    float s = 0.0f;
    for (int c = 0; c < HID; ++c) s += hr[c] * p[c];
    score[n] = tanhf(s / nrm[0]);
}
__global__ void topk_pool_kernel(const float* __restrict__ Hn, const float* __restrict__ score,
                                 float* __restrict__ pooled) {
    __shared__ float sv[512];
    __shared__ int   si[512];
    __shared__ int   selIdx[KTOP];
    __shared__ float selVal[KTOP];
    int g = blockIdx.x, t = threadIdx.x;
    float s = score[g * NPG + t];
    for (int k = 0; k < KTOP; ++k) {
        sv[t] = s; si[t] = t;
        __syncthreads();
        for (int o = 256; o > 0; o >>= 1) {
            if (t < o) {
                float s2 = sv[t + o]; int i2 = si[t + o];
                if (s2 > sv[t] || (s2 == sv[t] && i2 < si[t])) { sv[t] = s2; si[t] = i2; }
            }
            __syncthreads();
        }
        int winner = si[0];
        if (t == 0) { selIdx[k] = si[0]; selVal[k] = sv[0]; }
        if (t == winner) s = -2.0f;       // scores are in (-1,1)
        __syncthreads();
    }
    for (int i = t; i < KTOP * HID; i += 512) {
        int k = i >> 7, c = i & 127;
        pooled[(size_t)g * (KTOP * HID) + i] = Hn[(size_t)(g * NPG + selIdx[k]) * HID + c] * selVal[k];
    }
}

// ---------------- MLP head (tiny, plain f32) ----------------
__global__ void mlp0_kernel(const float* __restrict__ pooled, const float* __restrict__ w,
                            const float* __restrict__ b, float* __restrict__ h0) {
    int t = blockIdx.x * blockDim.x + threadIdx.x;
    if (t >= GG * HID) return;
    int g = t >> 7, o = t & 127;
    const float* pr = pooled + (size_t)g * (KTOP * HID);
    float acc = b[o];
    for (int i = 0; i < KTOP * HID; ++i) acc += pr[i] * w[(size_t)i * HID + o];
    h0[t] = fmaxf(acc, 0.0f);
}
__global__ void mlp1_kernel(const float* __restrict__ h0, const float* __restrict__ w,
                            const float* __restrict__ b, float* __restrict__ zf,
                            float* __restrict__ out_zf) {
    int t = blockIdx.x * blockDim.x + threadIdx.x;
    if (t >= GG * HID) return;
    int g = t >> 7, o = t & 127;
    const float* hr = h0 + (size_t)g * HID;
    float acc = b[o];
    for (int i = 0; i < HID; ++i) acc += hr[i] * w[(size_t)i * HID + o];
    float v = fmaxf(acc, 0.0f);
    zf[t] = v;
    out_zf[t] = v;
}
__global__ void mlp2_kernel(const float* __restrict__ zf, const float* __restrict__ w,
                            const float* __restrict__ b, float* __restrict__ logits) {
    int t = blockIdx.x * blockDim.x + threadIdx.x;
    if (t >= GG * OUTC) return;
    int g = t >> 1, o = t & 1;
    const float* zr = zf + (size_t)g * HID;
    float acc = b[o];
    for (int i = 0; i < HID; ++i) acc += zr[i] * w[(size_t)i * OUTC + o];
    logits[t] = acc;
}

// ---------------- host side ----------------
static inline size_t align256(size_t x) { return (x + 255) & ~(size_t)255; }

extern "C" void kernel_launch(void* const* d_in, const int* in_sizes, int n_in,
                              void* d_out, int out_size, void* d_ws, size_t ws_size,
                              hipStream_t stream) {
    (void)in_sizes; (void)n_in; (void)out_size; (void)ws_size;
    const float* x       = (const float*)d_in[0];
    const int*   esrc    = (const int*)d_in[1];
    const int*   edst    = (const int*)d_in[1] + EE;
    const float* W[3]    = { (const float*)d_in[3], (const float*)d_in[7], (const float*)d_in[11] };
    const float* AS[3]   = { (const float*)d_in[4], (const float*)d_in[8], (const float*)d_in[12] };
    const float* AD[3]   = { (const float*)d_in[5], (const float*)d_in[9], (const float*)d_in[13] };
    const float* BI[3]   = { (const float*)d_in[6], (const float*)d_in[10], (const float*)d_in[14] };
    const float* lnw[3]  = { (const float*)d_in[15], (const float*)d_in[17], (const float*)d_in[19] };
    const float* lnb[3]  = { (const float*)d_in[16], (const float*)d_in[18], (const float*)d_in[20] };
    const float* pool_p  = (const float*)d_in[21];
    const float* l0w = (const float*)d_in[22]; const float* l0b = (const float*)d_in[23];
    const float* l1w = (const float*)d_in[24]; const float* l1b = (const float*)d_in[25];
    const float* l2w = (const float*)d_in[26]; const float* l2b = (const float*)d_in[27];
    float* out = (float*)d_out;                 // [0,128): logits ; [128,8320): zf

    // workspace carve-out
    char* base = (char*)d_ws; size_t off = 0;
    auto ALLOC = [&](size_t bytes) { void* r = base + off; off += align256(bytes); return r; };
    float*    bufH   = (float*)ALLOC((size_t)NN * COUT * 4);
    float*    bufA   = (float*)ALLOC((size_t)NN * COUT * 4);
    __bf16*   Xhi    = (__bf16*)ALLOC((size_t)NN * COUT * 2);
    __bf16*   Xlo    = (__bf16*)ALLOC((size_t)NN * COUT * 2);
    __bf16*   Wth    = (__bf16*)ALLOC((size_t)COUT * COUT * 2);
    __bf16*   Wtl    = (__bf16*)ALLOC((size_t)COUT * COUT * 2);
    float*    ssum   = (float*)ALLOC((size_t)NN * HEADS * 4);
    float*    dsum   = (float*)ALLOC((size_t)NN * HEADS * 4);
    unsigned* mkey   = (unsigned*)ALLOC((size_t)NN * HEADS * 4);
    float*    mfb    = (float*)ALLOC((size_t)NN * HEADS * 4);
    float*    denom  = (float*)ALLOC((size_t)NN * HEADS * 4);
    int*      counts = (int*)ALLOC((size_t)NN * 4);
    int*      offs   = (int*)ALLOC((size_t)(NN + 1) * 4);
    int*      cursor = (int*)ALLOC((size_t)NN * 4);
    int*      bsums  = (int*)ALLOC(64 * 4);
    int*      ssrc   = (int*)ALLOC((size_t)EE * 4);
    float*    scoreb = (float*)ALLOC((size_t)NN * 4);
    float*    pnrm   = (float*)ALLOC(256);
    float*    pooled = (float*)ALLOC((size_t)GG * KTOP * HID * 4);
    float*    h0buf  = (float*)ALLOC((size_t)GG * HID * 4);
    float*    zfbuf  = (float*)ALLOC((size_t)GG * HID * 4);

    const int TB = 256;

    // ---- CSR build (once; graph static across layers) ----
    zero_int_kernel<<<(NN + TB - 1) / TB, TB, 0, stream>>>(counts, NN);
    zero_int_kernel<<<(NN + TB - 1) / TB, TB, 0, stream>>>(cursor, NN);
    hist_kernel<<<EE / TB, TB, 0, stream>>>(edst, counts);
    scan1_kernel<<<NN / 1024, 1024, 0, stream>>>(counts, offs, bsums);
    scan2_kernel<<<1, 32, 0, stream>>>(bsums, NN / 1024);
    scan3_kernel<<<NN / 1024, 1024, 0, stream>>>(offs, bsums);
    scatter_kernel<<<EE / TB, TB, 0, stream>>>(esrc, edst, offs, cursor, ssrc);

    // ---- three GAT layers ----
    for (int l = 0; l < 3; ++l) {
        int Fin = (l == 0) ? F_IN : COUT;
        const float* Xin = (l == 0) ? x : bufA;
        int nelem = NN * Fin;
        split_kernel<<<(nelem + TB - 1) / TB, TB, 0, stream>>>(Xin, nelem, Xhi, Xlo);
        wsplit_kernel<<<(Fin * COUT + TB - 1) / TB, TB, 0, stream>>>(W[l], Fin, Wth, Wtl);
        // (N/16)*(COUT/16) waves * 32 lanes / 256 threads per block
        gemm_wmma_kernel<<<(NN / 16) * (COUT / 16) * 32 / TB, TB, 0, stream>>>(Xhi, Xlo, Wth, Wtl, bufH, Fin);

        scores_kernel<<<NN * HEADS / TB, TB, 0, stream>>>(bufH, AS[l], AD[l], ssum, dsum);
        node_init_max_kernel<<<NN * HEADS / TB, TB, 0, stream>>>(ssum, dsum, mkey);
        edge_max_kernel<<<EE / TB, TB, 0, stream>>>(esrc, edst, ssum, dsum, mkey);
        node_denom_kernel<<<NN * HEADS / TB, TB, 0, stream>>>(ssum, dsum, mkey, mfb, denom);
        edge_denom_kernel<<<EE / TB, TB, 0, stream>>>(esrc, edst, ssum, dsum, mfb, denom);

        int meanHeads = (l == 2) ? 1 : 0;
        aggregate_kernel<<<NN * 32 / TB, TB, 0, stream>>>(bufH, ssum, dsum, mfb, denom,
                                                          offs, ssrc, BI[l], bufA, meanHeads);
        int C = meanHeads ? HID : COUT;
        graph_ln_kernel<<<GG, 256, 0, stream>>>(bufA, lnw[l], lnb[l], C, meanHeads ? 0 : 1);
    }

    // ---- TopK pooling + MLP head ----
    pnorm_kernel<<<1, 128, 0, stream>>>(pool_p, pnrm);
    score_kernel<<<NN / TB, TB, 0, stream>>>(bufA, pool_p, pnrm, scoreb);
    topk_pool_kernel<<<GG, 512, 0, stream>>>(bufA, scoreb, pooled);
    mlp0_kernel<<<GG * HID / TB, TB, 0, stream>>>(pooled, l0w, l0b, h0buf);
    mlp1_kernel<<<GG * HID / TB, TB, 0, stream>>>(h0buf, l1w, l1b, zfbuf, out + GG * OUTC);
    mlp2_kernel<<<1, GG * OUTC, 0, stream>>>(zfbuf, l2w, l2b, out);
}
